// FlashMLP_14723147891334
// MI455X (gfx1250) — matrix operand order repaired and checked by reading, not compiled, not run on hardware
//
#include <hip/hip_runtime.h>
#include <hip/hip_bf16.h>
#include <stdint.h>

typedef __attribute__((ext_vector_type(16))) __bf16 v16bf;
typedef __attribute__((ext_vector_type(8)))  float  v8f;
typedef int v4i __attribute__((vector_size(16)));   // int __vector(4) for async-LDS builtin

#define H_  16
#define B_  8192
#define D_  256
#define E_  1024
#define BM  64            // rows of x per workgroup
#define EB  32            // E-block (fusion block)
#define NB  (E_/EB)       // 32 blocks
#define W1S 36            // u32 row stride, w1 LDS (144B rows: 16B aligned, conflict-free)
#define W2S 260           // u32 row stride, w2 LDS (1040B rows: 16B aligned, conflict-free)
#define HS  (EB + 8)      // bf16 row stride, h LDS (80B rows, 16B aligned)

#define AS1 __attribute__((address_space(1)))
#define AS3 __attribute__((address_space(3)))

#if defined(__has_builtin)
#if __has_builtin(__builtin_amdgcn_global_load_async_to_lds_b128) && \
    __has_builtin(__builtin_amdgcn_s_wait_asynccnt)
#define ASYNC_LDS 1
#endif
#endif
#ifndef ASYNC_LDS
#define ASYNC_LDS 0
#endif

__device__ __forceinline__ uint32_t pack_bf16x2(float a, float b) {
  unsigned short lo = __builtin_bit_cast(unsigned short, (__bf16)a);
  unsigned short hi = __builtin_bit_cast(unsigned short, (__bf16)b);
  return (uint32_t)lo | ((uint32_t)hi << 16);
}

// ---- Pre-pass: fp32 weights -> bf16x2-packed, B-matrix-friendly layout.
// w1p[h][kd2][e] = pack(w1[h][2*kd2][e],   w1[h][2*kd2+1][e])    (row stride E_)
// w2p[h][ke2][d] = pack(w2[h][2*ke2][d],   w2[h][2*ke2+1][d])    (row stride D_)
__global__ void pack_w1_kernel(const float* __restrict__ w1, uint32_t* __restrict__ w1p) {
  int idx = blockIdx.x * blockDim.x + threadIdx.x;          // H*(D/2)*E = 2097152
  int e   = idx & (E_ - 1);
  int kd2 = (idx >> 10) & (D_/2 - 1);
  int h   = idx >> 17;
  const float* base = w1 + ((size_t)h * D_ + 2 * kd2) * E_ + e;
  w1p[idx] = pack_bf16x2(base[0], base[E_]);
}
__global__ void pack_w2_kernel(const float* __restrict__ w2, uint32_t* __restrict__ w2p) {
  int idx = blockIdx.x * blockDim.x + threadIdx.x;          // H*(E/2)*D = 2097152
  int d   = idx & (D_ - 1);
  int ke2 = (idx >> 8) & (E_/2 - 1);
  int h   = idx >> 17;
  const float* base = w2 + ((size_t)h * E_ + 2 * ke2) * D_ + d;
  w2p[idx] = pack_bf16x2(base[0], base[D_]);
}

// ---- Stage one E-block of packed weights into (double-buffered) LDS.
__device__ __forceinline__ void stage_packed(const uint32_t* __restrict__ w1p_h,
                                             const uint32_t* __restrict__ w2p_h,
                                             uint32_t* lw1, uint32_t* lw2,
                                             int eblk, int tid) {
#pragma unroll
  for (int i = 0; i < 4; ++i) {               // 1024 B128 chunks each / 256 thr
    int q = tid + i * 256;
    {
      int kd2 = q >> 3, e = (q & 7) * 4;      // w1 block: 128 rows x 32 u32
      const uint32_t* g = w1p_h + (size_t)kd2 * E_ + (size_t)eblk * EB + e;
      uint32_t* l = lw1 + kd2 * W1S + e;
#if ASYNC_LDS
      __builtin_amdgcn_global_load_async_to_lds_b128((AS1 v4i*)g, (AS3 v4i*)l, 0, 0);
#else
      *(uint4*)l = *(const uint4*)g;
#endif
    }
    {
      int ke2 = q >> 6, d = (q & 63) * 4;     // w2 block: 16 rows x 256 u32
      const uint32_t* g = w2p_h + (size_t)(eblk * (EB/2) + ke2) * D_ + d;
      uint32_t* l = lw2 + ke2 * W2S + d;
#if ASYNC_LDS
      __builtin_amdgcn_global_load_async_to_lds_b128((AS1 v4i*)g, (AS3 v4i*)l, 0, 0);
#else
      *(uint4*)l = *(const uint4*)g;
#endif
    }
  }
}

// ---- Fallback staging straight from fp32 weights (no workspace available).
__device__ __forceinline__ void stage_fp32(const float* __restrict__ w1g,
                                           const float* __restrict__ w2g,
                                           uint32_t* lw1, uint32_t* lw2,
                                           int eblk, int tid) {
#pragma unroll
  for (int i = 0; i < 16; ++i) {
    int p = tid + i * 256;
    int kd2 = p >> 5, e = p & 31;
    const float* q = w1g + (size_t)(2 * kd2) * E_ + (size_t)eblk * EB + e;
    lw1[kd2 * W1S + e] = pack_bf16x2(q[0], q[E_]);
  }
#pragma unroll
  for (int i = 0; i < 16; ++i) {
    int p = tid + i * 256;
    int ke2 = p >> 8, d = p & 255;
    const float* q = w2g + (size_t)(eblk * EB + 2 * ke2) * D_ + d;
    lw2[ke2 * W2S + d] = pack_bf16x2(q[0], q[D_]);
  }
  if (eblk + 1 < NB) {
    __builtin_prefetch(w1g + (size_t)tid * E_ + (size_t)(eblk + 1) * EB, 0, 3);
    __builtin_prefetch(w2g + (size_t)(eblk + 1) * EB * D_ + (size_t)tid * 32, 0, 3);
  }
}

template <bool PACKED>
__global__ void __launch_bounds__(256)
flashmlp_bf16_wmma_kernel(const float* __restrict__ x,
                          const void* __restrict__ w1v,
                          const void* __restrict__ w2v,
                          float* __restrict__ out) {
  // Double-buffered weight staging + silu(h) tile. ~73.5 KB of 320 KB WGP LDS.
  __shared__ __align__(16) uint32_t       lds_w1[2][(D_/2) * W1S];
  __shared__ __align__(16) uint32_t       lds_w2[2][(EB/2) * W2S];
  __shared__ __align__(16) unsigned short lds_h [BM * HS];

  const int tid  = threadIdx.x;
  const int lane = tid & 31;
  const int wave = tid >> 5;          // 8 waves
  const int l15  = lane & 15;
  const int kh   = lane >> 4;         // K-half select (wave32 WMMA layout)
  const int head = blockIdx.y;
  const int b0   = blockIdx.x * BM;
  const int mz   = wave & 3;          // GEMM1: this wave's M tile (z)
  const int nz   = wave >> 2;         // GEMM1: this wave's N tile (z), 0..1

  const uint32_t* w1p_h = PACKED ? (const uint32_t*)w1v + (size_t)head * (D_/2) * E_ : nullptr;
  const uint32_t* w2p_h = PACKED ? (const uint32_t*)w2v + (size_t)head * (E_/2) * D_ : nullptr;
  const float*    w1g   = PACKED ? nullptr : (const float*)w1v + (size_t)head * D_ * E_;
  const float*    w2g   = PACKED ? nullptr : (const float*)w2v + (size_t)head * E_ * D_;

  // ---- Preload this wave's x A-fragments (16 rows x 256 K) as bf16, once.
  v16bf xa[8];
  {
    const float* xr = x + ((size_t)head * B_ + (size_t)(b0 + mz * 16 + l15)) * D_;
#pragma unroll
    for (int ks = 0; ks < 8; ++ks) {
      const float* p0 = xr + ks * 32 + kh * 16;
      float4 a0 = *(const float4*)(p0);
      float4 a1 = *(const float4*)(p0 + 4);
      float4 c0 = *(const float4*)(p0 + 8);
      float4 c1 = *(const float4*)(p0 + 12);
      v16bf t;
      t[0]  = (__bf16)a0.x; t[1]  = (__bf16)a0.y; t[2]  = (__bf16)a0.z; t[3]  = (__bf16)a0.w;
      t[4]  = (__bf16)a1.x; t[5]  = (__bf16)a1.y; t[6]  = (__bf16)a1.z; t[7]  = (__bf16)a1.w;
      t[8]  = (__bf16)c0.x; t[9]  = (__bf16)c0.y; t[10] = (__bf16)c0.z; t[11] = (__bf16)c0.w;
      t[12] = (__bf16)c1.x; t[13] = (__bf16)c1.y; t[14] = (__bf16)c1.z; t[15] = (__bf16)c1.w;
      xa[ks] = t;
    }
  }

  const v8f vzero = {0.f, 0.f, 0.f, 0.f, 0.f, 0.f, 0.f, 0.f};
  v8f oacc[4][2];
#pragma unroll
  for (int m = 0; m < 4; ++m) { oacc[m][0] = vzero; oacc[m][1] = vzero; }

  // ---- Prologue: stage block 0 into buffer 0.
  if (PACKED) stage_packed(w1p_h, w2p_h, lds_w1[0], lds_w2[0], 0, tid);
  else        stage_fp32 (w1g,   w2g,   lds_w1[0], lds_w2[0], 0, tid);
#if ASYNC_LDS
  if (PACKED) __builtin_amdgcn_s_wait_asynccnt(0);
#endif
  __syncthreads();

  int buf = 0;
  for (int eblk = 0; eblk < NB; ++eblk) {
    // ---- Issue staging of the next block into the other buffer (overlaps compute).
    if (eblk + 1 < NB) {
      if (PACKED) stage_packed(w1p_h, w2p_h, lds_w1[buf ^ 1], lds_w2[buf ^ 1], eblk + 1, tid);
      else        stage_fp32 (w1g,   w2g,   lds_w1[buf ^ 1], lds_w2[buf ^ 1], eblk + 1, tid);
    }

    const uint32_t* lw1 = lds_w1[buf];
    const uint32_t* lw2 = lds_w2[buf];

    // ---- GEMM1: z(16x16 tile) = x_tile(mz) @ w1_blk(:, nz)   K = 256
    v8f z = vzero;
#pragma unroll
    for (int ks = 0; ks < 8; ++ks) {
      union { v16bf v; uint32_t u[8]; } bf;
#pragma unroll
      for (int v = 0; v < 8; ++v)
        bf.u[v] = lw1[(ks * 16 + kh * 8 + v) * W1S + nz * 16 + l15];
      z = __builtin_amdgcn_wmma_f32_16x16x32_bf16(false, xa[ks], false, bf.v,
                                                  (short)0, z, false, false);
    }

    // ---- silu -> h tile (bf16) in LDS (C layout: VGPR r -> M=r(+8))
    {
      int rb = mz * 16 + kh * 8;
      int cb = nz * 16 + l15;
#pragma unroll
      for (int r = 0; r < 8; ++r) {
        float zv = z[r];
        float s  = zv / (1.f + __expf(-zv));          // silu
        lds_h[(rb + r) * HS + cb] = __builtin_bit_cast(unsigned short, (__bf16)s);
      }
    }
    __syncthreads();

    // ---- GEMM2: out(64x256) += h(64x32) @ w2_blk(32x256)   K = 32
    v16bf ha[4];
#pragma unroll
    for (int m = 0; m < 4; ++m) {
      const unsigned short* hp = &lds_h[(m * 16 + l15) * HS + kh * 16];
      union { v16bf v; uint4 q[2]; } u;
      u.q[0] = *(const uint4*)(hp);
      u.q[1] = *(const uint4*)(hp + 8);
      ha[m] = u.v;
    }
#pragma unroll
    for (int j = 0; j < 2; ++j) {
      union { v16bf v; uint32_t u[8]; } bw;
      const int cb = wave * 32 + j * 16 + l15;
#pragma unroll
      for (int v = 0; v < 8; ++v)
        bw.u[v] = lw2[(kh * 8 + v) * W2S + cb];
#pragma unroll
      for (int m = 0; m < 4; ++m)
        oacc[m][j] = __builtin_amdgcn_wmma_f32_16x16x32_bf16(false, ha[m], false, bw.v,
                                                             (short)0, oacc[m][j], false, false);
    }

#if ASYNC_LDS
    if (PACKED) __builtin_amdgcn_s_wait_asynccnt(0);   // next buffer fully staged
#endif
    __syncthreads();   // everyone done reading `buf` and next buffer visible
    buf ^= 1;
  }

  // ---- Epilogue: fp32 store, coalesced across lanes (lane -> consecutive d)
#pragma unroll
  for (int m = 0; m < 4; ++m) {
#pragma unroll
    for (int j = 0; j < 2; ++j) {
      const int col = wave * 32 + j * 16 + l15;
      float* op = out + ((size_t)head * B_ + (size_t)(b0 + m * 16 + kh * 8)) * D_ + col;
#pragma unroll
      for (int r = 0; r < 8; ++r)
        op[(size_t)r * D_] = oacc[m][j][r];
    }
  }
}

extern "C" void kernel_launch(void* const* d_in, const int* in_sizes, int n_in,
                              void* d_out, int out_size, void* d_ws, size_t ws_size,
                              hipStream_t stream) {
  (void)in_sizes; (void)n_in; (void)out_size;
  const float* x  = (const float*)d_in[0];
  const float* w1 = (const float*)d_in[1];
  const float* w2 = (const float*)d_in[2];
  float* out = (float*)d_out;

  dim3 grid(B_ / BM, H_);   // (128, 16) workgroups
  dim3 block(256);          // 8 wave32 waves

  const size_t n1 = (size_t)H_ * (D_/2) * E_;   // u32 elements
  const size_t n2 = (size_t)H_ * (E_/2) * D_;
  if (ws_size >= (n1 + n2) * sizeof(uint32_t)) {
    uint32_t* w1p = (uint32_t*)d_ws;
    uint32_t* w2p = w1p + n1;
    pack_w1_kernel<<<(int)(n1 / 256), 256, 0, stream>>>(w1, w1p);
    pack_w2_kernel<<<(int)(n2 / 256), 256, 0, stream>>>(w2, w2p);
    flashmlp_bf16_wmma_kernel<true><<<grid, block, 0, stream>>>(x, w1p, w2p, out);
  } else {
    flashmlp_bf16_wmma_kernel<false><<<grid, block, 0, stream>>>(x, w1, w2, out);
  }
}
